// value_weight_25744033972900
// MI455X (gfx1250) — matrix operand order repaired and checked by reading, not compiled, not run on hardware
//
#include <hip/hip_runtime.h>
#include <hip/hip_bf16.h>

typedef __bf16 bf16t;
typedef bf16t v16bf __attribute__((ext_vector_type(16)));
typedef float v8f   __attribute__((ext_vector_type(8)));

__device__ __forceinline__ unsigned short f32_to_bf16_rne(float f) {
    unsigned u = __float_as_uint(f);
    u += 0x7fffu + ((u >> 16) & 1u);
    return (unsigned short)(u >> 16);
}
__device__ __forceinline__ float bf16_to_f32(unsigned short h) {
    return __uint_as_float((unsigned)h << 16);
}

// ---------------------------------------------------------------------------
// Weight pre-pass: Wt[n][k] = bf16(W[k][n]), zero-padded to Kp (Kp % 32 == 0).
// ---------------------------------------------------------------------------
__global__ void wtrans_kernel(const float* __restrict__ W, unsigned short* __restrict__ Wt,
                              int K, int N, int Kp, long total)
{
    long i = (long)blockIdx.x * blockDim.x + threadIdx.x;
    if (i >= total) return;
    int k = (int)(i % Kp);
    long n = i / Kp;
    float v = (k < K) ? W[(long)k * N + n] : 0.f;
    Wt[i] = f32_to_bf16_rne(v);
}

// elementwise f32 -> bf16
__global__ void f2bf_kernel(const float* __restrict__ in, unsigned short* __restrict__ out,
                            long total)
{
    long i = (long)blockIdx.x * blockDim.x + threadIdx.x;
    if (i >= total) return;
    out[i] = f32_to_bf16_rne(in[i]);
}

// ---------------------------------------------------------------------------
// GEMM: C[M,N](f32) = A[M,K](bf16) * Wt^T(bf16) + bias.  Wt is [N][K]
// pre-transposed, zero-padded.  Preconditions: M%16==0, N%32==0, K%32==0.
// One wave -> 16x32 output (two WMMA tiles sharing the A fragment).
// Loop body: 6 aligned b128 loads + 2 v_wmma, no conversion ALU, full EXEC.
// ---------------------------------------------------------------------------
__global__ __launch_bounds__(256) void gemm_wmma_kernel(
    const unsigned short* __restrict__ A, const unsigned short* __restrict__ Wt,
    const float* __restrict__ bias, float* __restrict__ C,
    int M, int N, int K)
{
    int wave    = blockIdx.x * (blockDim.x >> 5) + (threadIdx.x >> 5);
    int tilesN2 = N >> 5;
    int tilesM  = M >> 4;
    if (wave >= tilesM * tilesN2) return;   // whole-wave exit, EXEC stays full
    int tm   = wave / tilesN2;
    int tn   = wave % tilesN2;
    int lane = threadIdx.x & 31;
    int half = lane >> 4;
    int l15  = lane & 15;
    int row  = tm * 16 + l15;
    int col0 = tn * 32 + l15;
    int col1 = col0 + 16;

    const unsigned short* Arow = A + (long)row * K;
    const unsigned short* W0   = Wt + (long)col0 * K;
    const unsigned short* W1   = Wt + (long)col1 * K;

    v8f acc0 = {0.f, 0.f, 0.f, 0.f, 0.f, 0.f, 0.f, 0.f};
    v8f acc1 = {0.f, 0.f, 0.f, 0.f, 0.f, 0.f, 0.f, 0.f};

    for (int k0 = 0; k0 < K; k0 += 32) {
        __builtin_prefetch(Arow + k0 + 128, 0, 1);
        // A: element e -> K = k0 + half*16 + e (contiguous)
        union { v16bf v; uint4 q[2]; } a;
        a.q[0] = *(const uint4*)(Arow + k0 + half * 16);
        a.q[1] = *(const uint4*)(Arow + k0 + half * 16 + 8);
        // B: element e -> K = k0 + half*16 + e (contiguous in Wt row)
        union { v16bf v; uint4 q[2]; } b0, b1;
        const uint4* pb0 = (const uint4*)(W0 + k0 + half * 16);
        const uint4* pb1 = (const uint4*)(W1 + k0 + half * 16);
        b0.q[0] = pb0[0]; b0.q[1] = pb0[1];
        b1.q[0] = pb1[0]; b1.q[1] = pb1[1];

        acc0 = __builtin_amdgcn_wmma_f32_16x16x32_bf16(
            false, a.v, false, b0.v, (short)0, acc0, false, false);
        acc1 = __builtin_amdgcn_wmma_f32_16x16x32_bf16(
            false, a.v, false, b1.v, (short)0, acc1, false, false);
    }

    union { v8f v; float f[8]; } r0, r1;
    r0.v = acc0; r1.v = acc1;
    float bc0 = bias[col0], bc1 = bias[col1];
#pragma unroll
    for (int rr = 0; rr < 8; ++rr) {
        int m = tm * 16 + rr + half * 8;
        C[(long)m * N + col0] = r0.f[rr] + bc0;
        C[(long)m * N + col1] = r1.f[rr] + bc1;
    }
}

// conv3: ragged N==3 output head (memory-bound; plain VALU kernel, bf16 input)
__global__ void conv3_kernel(const unsigned short* __restrict__ A, const float* __restrict__ W,
                             const float* __restrict__ b, float* __restrict__ out,
                             int M, int K)
{
    int t = blockIdx.x * blockDim.x + threadIdx.x;
    if (t >= M) return;
    const unsigned short* a = A + (long)t * K;
    float s0 = b[0], s1 = b[1], s2 = b[2];
    for (int k = 0; k < K; ++k) {
        float v = bf16_to_f32(a[k]);
        s0 += v * W[k * 3 + 0];
        s1 += v * W[k * 3 + 1];
        s2 += v * W[k * 3 + 2];
    }
    out[t * 3 + 0] = s0;
    out[t * 3 + 1] = s1;
    out[t * 3 + 2] = s2;
}

// ---------------------------------------------------------------------------
// BatchNorm statistics on the f32 GEMM output: one block per channel.
// ---------------------------------------------------------------------------
__global__ void bn_stats_kernel(const float* __restrict__ X,
                                float* __restrict__ mean, float* __restrict__ var,
                                int M, int N)
{
    int n = blockIdx.x;
    __shared__ float ss[256], sq[256];
    float s = 0.f, q = 0.f;
    for (int m = threadIdx.x; m < M; m += blockDim.x) {
        float v = X[(long)m * N + n];
        s += v; q += v * v;
    }
    ss[threadIdx.x] = s; sq[threadIdx.x] = q;
    __syncthreads();
    for (int st = 128; st > 0; st >>= 1) {
        if (threadIdx.x < st) {
            ss[threadIdx.x] += ss[threadIdx.x + st];
            sq[threadIdx.x] += sq[threadIdx.x + st];
        }
        __syncthreads();
    }
    if (threadIdx.x == 0) {
        float mu = ss[0] / (float)M;
        mean[n] = mu;
        var[n]  = fmaxf(sq[0] / (float)M - mu * mu, 0.f);
    }
}

// normalize + relu in f32, emit bf16 operand for next GEMM
__global__ void bn_relu_kernel(const float* __restrict__ Y, unsigned short* __restrict__ Xo,
                               const float* __restrict__ mean, const float* __restrict__ var,
                               const float* __restrict__ g, const float* __restrict__ beta,
                               long total, int N)
{
    long i = (long)blockIdx.x * blockDim.x + threadIdx.x;
    if (i >= total) return;
    int n = (int)(i % N);
    float x = Y[i];
    x = g[n] * (x - mean[n]) * rsqrtf(var[n] + 1e-5f) + beta[n];
    Xo[i] = f32_to_bf16_rne(fmaxf(x, 0.f));
}

// ---------------------------------------------------------------------------
// Farthest point sampling: one workgroup per batch, dist[] resident in LDS.
// ---------------------------------------------------------------------------
__global__ __launch_bounds__(512) void fps_kernel(
    const float* __restrict__ xyz, int* __restrict__ out, int N, int npoint)
{
    int b = blockIdx.x;
    __shared__ float dist[8192];
    __shared__ float rv[512];
    __shared__ int   ri[512];
    const float* p = xyz + (long)b * N * 3;
    int tid = threadIdx.x, bd = blockDim.x;
    for (int i = tid; i < N; i += bd) dist[i] = 1e10f;
    __syncthreads();
    int far = 0;
    for (int it = 0; it < npoint; ++it) {
        if (tid == 0) out[b * npoint + it] = far;
        float cx = p[far * 3 + 0], cy = p[far * 3 + 1], cz = p[far * 3 + 2];
        float best = -1.f; int bi = 0;
        for (int i = tid; i < N; i += bd) {
            float dx = p[i * 3 + 0] - cx;
            float dy = p[i * 3 + 1] - cy;
            float dz = p[i * 3 + 2] - cz;
            float d  = dx * dx + dy * dy + dz * dz;
            float nd = fminf(dist[i], d);
            dist[i] = nd;
            if (nd > best) { best = nd; bi = i; }
        }
        rv[tid] = best; ri[tid] = bi;
        __syncthreads();
        for (int st = bd >> 1; st > 0; st >>= 1) {
            if (tid < st && rv[tid + st] > rv[tid]) {
                rv[tid] = rv[tid + st]; ri[tid] = ri[tid + st];
            }
            __syncthreads();
        }
        far = ri[0];
        __syncthreads();
    }
}

// dst[b,s,c] = src[b, idx[b,s], c]   (f32 geometry gather)
__global__ void gather_kernel(const float* __restrict__ src, const int* __restrict__ idx,
                              float* __restrict__ dst, int N, int S, int C, long total)
{
    long i = (long)blockIdx.x * blockDim.x + threadIdx.x;
    if (i >= total) return;
    int c = (int)(i % C);
    long t = i / C;
    int s = (int)(t % S);
    int b = (int)(t / S);
    dst[i] = src[((long)b * N + idx[b * S + s]) * C + c];
}

// First-nsample-in-index-order ball query (matches sort-then-slice reference).
__global__ void ball_query_kernel(const float* __restrict__ xyz,
                                  const float* __restrict__ nxyz,
                                  int* __restrict__ idx,
                                  int N, int S, float r2, int K, int BS)
{
    int t = blockIdx.x * blockDim.x + threadIdx.x;
    if (t >= BS) return;
    int b = t / S;
    const float* p = xyz + (long)b * N * 3;
    float cx = nxyz[t * 3 + 0], cy = nxyz[t * 3 + 1], cz = nxyz[t * 3 + 2];
    int* o = idx + (long)t * K;
    int cnt = 0;
    for (int j = 0; j < N && cnt < K; ++j) {
        float dx = p[j * 3 + 0] - cx;
        float dy = p[j * 3 + 1] - cy;
        float dz = p[j * 3 + 2] - cz;
        if (dx * dx + dy * dy + dz * dz <= r2) o[cnt++] = j;
    }
    int f = (cnt > 0) ? o[0] : 0;
    for (int k = cnt; k < K; ++k) o[k] = f;
}

// grouped[b,s,k,:] = bf16([xyz[idx]-new_xyz , points[idx], 0-pad to Cpad])
__global__ void group_kernel(const float* __restrict__ xyz, const unsigned short* __restrict__ pts,
                             const int* __restrict__ idx, const float* __restrict__ nxyz,
                             unsigned short* __restrict__ out, int N, int S, int K, int Cp,
                             int Cpad, int BSK)
{
    int t = blockIdx.x * blockDim.x + threadIdx.x;
    if (t >= BSK) return;
    int bs = t / K;
    int b = bs / S;
    int j = idx[t];
    unsigned short* o = out + (long)t * Cpad;
    const float* pj = xyz + ((long)b * N + j) * 3;
    const float* c  = nxyz + (long)bs * 3;
    o[0] = f32_to_bf16_rne(pj[0] - c[0]);
    o[1] = f32_to_bf16_rne(pj[1] - c[1]);
    o[2] = f32_to_bf16_rne(pj[2] - c[2]);
    const unsigned short* f = pts + ((long)b * N + j) * Cp;
    for (int cc = 0; cc < Cp; ++cc) o[3 + cc] = f[cc];
    for (int cc = 3 + Cp; cc < Cpad; ++cc) o[cc] = 0;
}

// out[bs,c] = max_k X[bs,k,c]   (bf16 in/out; exact since max picks an input)
__global__ void maxpool_kernel(const unsigned short* __restrict__ X,
                               unsigned short* __restrict__ out,
                               int K, int C, long total)
{
    long i = (long)blockIdx.x * blockDim.x + threadIdx.x;
    if (i >= total) return;
    int c = (int)(i % C);
    long bs = i / C;
    const unsigned short* p = X + bs * (long)K * C + c;
    float m = bf16_to_f32(p[0]);
    unsigned short mb = p[0];
    for (int k = 1; k < K; ++k) {
        unsigned short hb = p[(long)k * C];
        float h = bf16_to_f32(hb);
        if (h > m) { m = h; mb = hb; }
    }
    out[i] = mb;
}

// 3-NN with inverse-distance weights (f32 geometry)
__global__ void knn3_kernel(const float* __restrict__ xyz1, const float* __restrict__ xyz2,
                            int* __restrict__ idx3, float* __restrict__ w3,
                            int N1, int N2, int BN1)
{
    int t = blockIdx.x * blockDim.x + threadIdx.x;
    if (t >= BN1) return;
    int b = t / N1;
    float cx = xyz1[t * 3 + 0], cy = xyz1[t * 3 + 1], cz = xyz1[t * 3 + 2];
    const float* p = xyz2 + (long)b * N2 * 3;
    float d0 = 1e30f, d1 = 1e30f, d2v = 1e30f;
    int i0 = 0, i1 = 0, i2 = 0;
    for (int j = 0; j < N2; ++j) {
        float dx = p[j * 3 + 0] - cx;
        float dy = p[j * 3 + 1] - cy;
        float dz = p[j * 3 + 2] - cz;
        float d = dx * dx + dy * dy + dz * dz;
        if (d < d0)      { d2v = d1; i2 = i1; d1 = d0; i1 = i0; d0 = d; i0 = j; }
        else if (d < d1) { d2v = d1; i2 = i1; d1 = d;  i1 = j; }
        else if (d < d2v){ d2v = d;  i2 = j; }
    }
    float w0 = 1.f / (d0 + 1e-8f);
    float w1 = 1.f / (d1 + 1e-8f);
    float w2 = 1.f / (d2v + 1e-8f);
    float s = w0 + w1 + w2;
    idx3[t * 3 + 0] = i0; idx3[t * 3 + 1] = i1; idx3[t * 3 + 2] = i2;
    w3[t * 3 + 0] = w0 / s; w3[t * 3 + 1] = w1 / s; w3[t * 3 + 2] = w2 / s;
}

// out[b,n,:] = bf16([pts1[b,n,:], sum_k w_k * pts2[b, idx_k, :]])
__global__ void interp_concat_kernel(const unsigned short* __restrict__ pts1,
                                     const unsigned short* __restrict__ pts2,
                                     const int* __restrict__ idx3, const float* __restrict__ w3,
                                     unsigned short* __restrict__ out,
                                     int N1, int C1, int C2, int N2, long total)
{
    long i = (long)blockIdx.x * blockDim.x + threadIdx.x;
    if (i >= total) return;
    int Ct = C1 + C2;
    int c = (int)(i % Ct);
    long t = i / Ct;
    int b = (int)(t / N1);
    if (c < C1) {
        out[i] = pts1[t * C1 + c];
    } else {
        int cc = c - C1;
        const unsigned short* p2 = pts2 + (long)b * N2 * C2;
        float acc = 0.f;
#pragma unroll
        for (int k = 0; k < 3; ++k)
            acc += bf16_to_f32(p2[(long)idx3[t * 3 + k] * C2 + cc]) * w3[t * 3 + k];
        out[i] = f32_to_bf16_rne(acc);
    }
}

// top-25 soft projection (f32 geometry)
__global__ void project_kernel(const float* __restrict__ x, const float* __restrict__ sim,
                               const float* __restrict__ temp, float* __restrict__ proj,
                               int N, int S, int BS)
{
    int t = blockIdx.x * blockDim.x + threadIdx.x;
    if (t >= BS) return;
    int b = t / S;
    float cx = sim[t * 3 + 0], cy = sim[t * 3 + 1], cz = sim[t * 3 + 2];
    const float* p = x + (long)b * N * 3;
    const int G = 25;
    float kd[G]; int ki[G];
#pragma unroll
    for (int g = 0; g < G; ++g) { kd[g] = 1e30f; ki[g] = 0; }
    for (int j = 0; j < N; ++j) {
        float dx = p[j * 3 + 0] - cx;
        float dy = p[j * 3 + 1] - cy;
        float dz = p[j * 3 + 2] - cz;
        float d = dx * dx + dy * dy + dz * dz;
        if (d < kd[G - 1]) {
            int pos = G - 1;
            while (pos > 0 && kd[pos - 1] > d) {
                kd[pos] = kd[pos - 1]; ki[pos] = ki[pos - 1]; --pos;
            }
            kd[pos] = d; ki[pos] = j;
        }
    }
    float tt = temp[0];
    float sigma = fmaxf(tt * tt, 1e-4f);
    float inv = 1.f / (sigma + 1e-8f);
    float mn = kd[0];
    float wsum = 0.f, px = 0.f, py = 0.f, pz = 0.f;
    for (int g = 0; g < G; ++g) {
        float w = __expf(-(kd[g] - mn) * inv);
        int j = ki[g];
        wsum += w;
        px += w * p[j * 3 + 0];
        py += w * p[j * 3 + 1];
        pz += w * p[j * 3 + 2];
    }
    proj[t * 3 + 0] = px / wsum;
    proj[t * 3 + 1] = py / wsum;
    proj[t * 3 + 2] = pz / wsum;
}

// ===========================================================================
extern "C" void kernel_launch(void* const* d_in, const int* in_sizes, int n_in,
                              void* d_out, int out_size, void* d_ws, size_t ws_size,
                              hipStream_t stream)
{
    (void)in_sizes; (void)n_in; (void)ws_size;
    const int B = 2, N0 = 8192;
    const float* x = (const float*)d_in[0];
    const int S_out = out_size / (B * 3 * 2);   // num_samples (2048)

    auto P = [&](int i) { return (const float*)d_in[i]; };
    const int CONV3W = 2, CONV3B = 3, FP1 = 4, FP2 = 16, FP3 = 24, FP4 = 32,
              HEAD = 40, SA1 = 48, SA2 = 60, SA3 = 72, SA4 = 84, TEMP = 96;

    // ---- workspace arena ----
    char* basep = (char*)d_ws;
    size_t off = 0;
    auto alloc = [&](size_t bytes) -> void* {
        void* p = basep + off;
        off += (bytes + 255) & ~(size_t)255;
        return p;
    };
    typedef unsigned short ush;
    const size_t BIG = (size_t)B * 8192 * 1024;          // max M*N elements
    float* ybuf = (float*)alloc(BIG * 4);                 // f32 GEMM output (reused)
    ush* bfA = (ush*)alloc(BIG * 2);                      // bf16 activation ping
    ush* bfB = (ush*)alloc(BIG * 2);                      // bf16 activation pong
    float* l1x = (float*)alloc((size_t)B * 4096 * 3 * 4);
    float* l2x = (float*)alloc((size_t)B * 2048 * 3 * 4);
    float* l3x = (float*)alloc((size_t)B * 1024 * 3 * 4);
    float* l4x = (float*)alloc((size_t)B * 256 * 3 * 4);
    ush* l1p = (ush*)alloc((size_t)B * 4096 * 64 * 2);
    ush* l2p = (ush*)alloc((size_t)B * 2048 * 128 * 2);
    ush* l3p = (ush*)alloc((size_t)B * 1024 * 256 * 2);
    ush* l4p = (ush*)alloc((size_t)B * 256 * 512 * 2);
    ush* cat4 = (ush*)alloc((size_t)B * 1024 * 768 * 2);
    ush* cat3 = (ush*)alloc((size_t)B * 2048 * 640 * 2);
    ush* cat2 = (ush*)alloc((size_t)B * 4096 * 1088 * 2);
    ush* xyzbf = (ush*)alloc((size_t)B * 8192 * 3 * 2);
    float* hbuf = (float*)alloc((size_t)B * 8192 * 3 * 4);
    int*   fpsidx = (int*)alloc((size_t)B * 4096 * 4);
    int*   grpidx = (int*)alloc((size_t)B * 4096 * 32 * 4);
    int*   knnidx = (int*)alloc((size_t)B * 8192 * 3 * 4);
    float* knnw   = (float*)alloc((size_t)B * 8192 * 3 * 4);
    float* meanb  = (float*)alloc(1024 * 4);
    float* varb   = (float*)alloc(1024 * 4);
    ush* wtbuf = (ush*)alloc((size_t)1024 * 1024 * 2);    // 1024*1024 elements; fp2 layer 0 has N*Kp = 1024*1088

    // ---- helpers ----
    auto gemm = [&](const ush* A, const float* W, const float* bias, float* C,
                    int M, int N, int K, int Kp) {
        long wt = (long)N * Kp;
        wtrans_kernel<<<(int)((wt + 255) / 256), 256, 0, stream>>>(W, wtbuf, K, N, Kp, wt);
        long waves = (long)(M >> 4) * (N >> 5);
        int blocks = (int)((waves + 7) / 8);
        gemm_wmma_kernel<<<blocks, 256, 0, stream>>>(A, wtbuf, bias, C, M, N, Kp);
    };
    // chain: X bf16 -> [gemm -> f32 ybuf -> bn_stats -> bn_relu -> bf16] x nL
    auto chain = [&](int pbase, int nL, const int* dims, int K0pad,
                     ush* X0, ush* t1, ush* t2, int M) -> ush* {
        const ush* cur = X0;
        ush* out = nullptr;
        for (int li = 0; li < nL; ++li) {
            out = (li & 1) ? t2 : t1;
            int K  = dims[li];
            int Kp = (li == 0) ? K0pad : K;
            int N  = dims[li + 1];
            gemm(cur, P(pbase + li * 4 + 0), P(pbase + li * 4 + 1), ybuf, M, N, K, Kp);
            bn_stats_kernel<<<N, 256, 0, stream>>>(ybuf, meanb, varb, M, N);
            long tot = (long)M * N;
            bn_relu_kernel<<<(int)((tot + 255) / 256), 256, 0, stream>>>(
                ybuf, out, meanb, varb, P(pbase + li * 4 + 3), P(pbase + li * 4 + 2), tot, N);
            cur = out;
        }
        return out;
    };
    auto sa = [&](const float* ixyz, const ush* ipts, int N, int S, float radius,
                  int Cp, int pbase, const int* dims, float* oxyz, ush* opts) {
        fps_kernel<<<B, 512, 0, stream>>>(ixyz, fpsidx, N, S);
        long gt = (long)B * S * 3;
        gather_kernel<<<(int)((gt + 255) / 256), 256, 0, stream>>>(ixyz, fpsidx, oxyz, N, S, 3, gt);
        int BS = B * S;
        ball_query_kernel<<<(BS + 255) / 256, 256, 0, stream>>>(
            ixyz, oxyz, grpidx, N, S, radius * radius, 32, BS);
        int BSK = BS * 32;
        int Cpad = (3 + Cp + 31) & ~31;
        group_kernel<<<(BSK + 255) / 256, 256, 0, stream>>>(
            ixyz, ipts, grpidx, oxyz, bfA, N, S, 32, Cp, Cpad, BSK);
        ush* feat = chain(pbase, 3, dims, Cpad, bfA, bfB, bfA, BSK);   // ends bfB
        int Cout = dims[3];
        long mt = (long)BS * Cout;
        maxpool_kernel<<<(int)((mt + 255) / 256), 256, 0, stream>>>(feat, opts, 32, Cout, mt);
    };
    auto fp = [&](const float* xyz1, const float* xyz2, const ush* pts1, const ush* pts2,
                  int N1, int N2, int C1, int C2, int pbase, int nL, const int* dims,
                  ush* cat, ush* t1, ush* t2) -> ush* {
        int BN1 = B * N1;
        knn3_kernel<<<(BN1 + 255) / 256, 256, 0, stream>>>(xyz1, xyz2, knnidx, knnw, N1, N2, BN1);
        long tot = (long)BN1 * (C1 + C2);
        interp_concat_kernel<<<(int)((tot + 255) / 256), 256, 0, stream>>>(
            pts1, pts2, knnidx, knnw, cat, N1, C1, C2, N2, tot);
        return chain(pbase, nL, dims, dims[0], cat, t1, t2, BN1);
    };

    static const int d_sa1[4] = {6, 32, 32, 64};
    static const int d_sa2[4] = {67, 64, 64, 128};
    static const int d_sa3[4] = {131, 128, 128, 256};
    static const int d_sa4[4] = {259, 256, 256, 512};
    static const int d_fp4[3] = {768, 256, 512};
    static const int d_fp3[3] = {640, 512, 1024};
    static const int d_fp2[3] = {1088, 1024, 1024};
    static const int d_fp1[4] = {1024, 1024, 1024, 1024};
    static const int d_head[3] = {1024, 1024, 512};

    // bf16 copy of xyz as sa1's "points" input
    {
        long tot = (long)B * N0 * 3;
        f2bf_kernel<<<(int)((tot + 255) / 256), 256, 0, stream>>>(x, xyzbf, tot);
    }

    // ---- encoder ----
    sa(x,   xyzbf, N0,  4096, 0.1f, 3,   SA1, d_sa1, l1x, l1p);
    sa(l1x, l1p, 4096, 2048, 0.2f, 64,  SA2, d_sa2, l2x, l2p);
    sa(l2x, l2p, 2048, 1024, 0.4f, 128, SA3, d_sa3, l3x, l3p);
    sa(l3x, l3p, 1024, 256,  0.8f, 256, SA4, d_sa4, l4x, l4p);

    // ---- decoder (2-layer chains end in bfB; pts2 consumed before clobber) ----
    ush* f4 = fp(l3x, l4x, l3p, l4p, 1024, 256, 256, 512, FP4, 2, d_fp4, cat4, bfA, bfB);
    ush* f3 = fp(l2x, l3x, l2p, f4, 2048, 1024, 128, 512, FP3, 2, d_fp3, cat3, bfA, bfB);
    ush* f2 = fp(l1x, l2x, l1p, f3, 4096, 2048, 64, 1024, FP2, 2, d_fp2, cat2, bfA, bfB);

    // fp1: concat lives in bfA (interp reads f2=bfB before chain clobbers it)
    {
        int BN1 = B * N0;
        knn3_kernel<<<(BN1 + 255) / 256, 256, 0, stream>>>(x, l1x, knnidx, knnw, N0, 4096, BN1);
        long tot = (long)BN1 * 1024;
        interp_concat_kernel<<<(int)((tot + 255) / 256), 256, 0, stream>>>(
            nullptr, f2, knnidx, knnw, bfA, N0, 0, 1024, 4096, tot);
    }
    ush* f1 = chain(FP1, 3, d_fp1, 1024, bfA, bfB, bfA, B * N0);   // ends bfB

    // ---- head ----
    ush* hfeat = chain(HEAD, 2, d_head, 1024, f1, bfA, bfB, B * N0); // ends bfB
    conv3_kernel<<<(B * N0 + 255) / 256, 256, 0, stream>>>(
        hfeat, P(CONV3W), P(CONV3B), hbuf, B * N0, 512);

    // ---- sampling + soft projection ----
    fps_kernel<<<B, 512, 0, stream>>>(hbuf, fpsidx, N0, S_out);
    float* simout = (float*)d_out;
    float* projout = simout + (long)B * S_out * 3;
    long gt = (long)B * S_out * 3;
    gather_kernel<<<(int)((gt + 255) / 256), 256, 0, stream>>>(hbuf, fpsidx, simout, N0, S_out, 3, gt);
    int BS = B * S_out;
    project_kernel<<<(BS + 255) / 256, 256, 0, stream>>>(x, simout, P(TEMP), projout, N0, S_out, BS);
}
